// DiffusionHungarianMatcher_27092653703624
// MI455X (gfx1250) — compile-verified, matrix operand order checked by reading, not run on hardware
//
#include <hip/hip_runtime.h>

// DETR greedy Hungarian-style matcher for MI455X (gfx1250, wave32).
// Kernel A: cost matrix; class cost done as one-hot WMMA GEMM (f16 hi/lo split
//           accumulated in f32 -> effectively exact), box L1 + GIoU in VALU.
// Kernel B: per-batch greedy argmin with incremental per-row minima cached in
//           LDS: each step reduces over 900 row-minima instead of 115200
//           elements, and only rows whose argmin column was consumed rescan.
//           C in d_out stays pristine, matching the reference's returned C.

typedef __attribute__((ext_vector_type(16))) _Float16 v16h;
typedef __attribute__((ext_vector_type(8)))  float    v8f;

#define QN 900          // queries
#define CN 91           // classes
#define KP 96           // classes padded to multiple of 32
#define TN 128          // targets
#define QTILE 16
#define QT_TILES ((QN + QTILE - 1) / QTILE)   // 57

__global__ __launch_bounds__(256) void detr_cost_kernel(
    const float* __restrict__ logits,   // [B,QN,CN]
    const float* __restrict__ pboxes,   // [B,QN,4] cxcywh
    const int*   __restrict__ labels,   // [B,TN]
    const float* __restrict__ tboxes,   // [B,TN,4] cxcywh
    float* __restrict__ Cout)           // [B,QN,TN]
{
    __shared__ _Float16 sHi[QTILE * KP];
    __shared__ _Float16 sLo[QTILE * KP];
    __shared__ float    sPB[QTILE * 4];

    const int b   = blockIdx.y;
    const int q0  = blockIdx.x * QTILE;
    const int tid = threadIdx.x;

    // ---- phase 1: softmax of 16 query rows into LDS as f16 hi/lo split ----
    if (tid < QTILE) {
        const int q = q0 + tid;
        if (q < QN) {
            const float* lrow = logits + ((size_t)b * QN + q) * CN;
            float m = -3.402823466e+38f;
            for (int c = 0; c < CN; ++c) m = fmaxf(m, lrow[c]);
            float s = 0.f;
            for (int c = 0; c < CN; ++c) s += expf(lrow[c] - m);
            const float inv = 1.f / s;
            for (int c = 0; c < KP; ++c) {
                float p = (c < CN) ? expf(lrow[c] - m) * inv : 0.f;
                _Float16 hi = (_Float16)p;
                sHi[tid * KP + c] = hi;
                sLo[tid * KP + c] = (_Float16)(p - (float)hi);
            }
            const float* pb = pboxes + ((size_t)b * QN + q) * 4;
            sPB[tid * 4 + 0] = pb[0];
            sPB[tid * 4 + 1] = pb[1];
            sPB[tid * 4 + 2] = pb[2];
            sPB[tid * 4 + 3] = pb[3];
        } else {
            for (int c = 0; c < KP; ++c) {
                sHi[tid * KP + c] = (_Float16)0.f;
                sLo[tid * KP + c] = (_Float16)0.f;
            }
            sPB[tid * 4 + 0] = 0.f; sPB[tid * 4 + 1] = 0.f;
            sPB[tid * 4 + 2] = 0.f; sPB[tid * 4 + 3] = 0.f;
        }
    }
    __syncthreads();

    // ---- phase 2: each wave owns one 16-wide t tile; class cost via WMMA ----
    const int lane = tid & 31;
    const int wave = tid >> 5;                 // 0..7 -> 8*16 = 128 = TN
    const int nlo  = lane & 15;
    const int hi8  = (lane >> 4) * 8;          // 0 for lanes 0-15, 8 for 16-31
    const int t    = wave * 16 + nlo;

    const int   label = labels[b * TN + t];
    const float* tb = tboxes + ((size_t)b * TN + t) * 4;
    const float tcx = tb[0], tcy = tb[1], tw = tb[2], th = tb[3];

    v8f acc = {};
    #pragma unroll
    for (int ks = 0; ks < 3; ++ks) {
        v16h aHi, aLo, bf;
        #pragma unroll
        for (int e = 0; e < 16; ++e) {
            const int v  = e >> 1, pp = e & 1;
            // 16-bit A layout (ISA 7.12.2): vgprs 0-3 -> K 0..7, 4-7 -> K 16..23,
            // +8 for lanes 16-31; row M = lane&15. B mirrored (column-major).
            const int kl = ((v < 4) ? (2 * v + pp) : (2 * v + 8 + pp)) + hi8;
            const int K  = ks * 32 + kl;
            aHi[e] = sHi[nlo * KP + K];
            aLo[e] = sLo[nlo * KP + K];
            bf[e]  = (K == label) ? (_Float16)1.0f : (_Float16)0.0f;
        }
        acc = __builtin_amdgcn_wmma_f32_16x16x32_f16(false, aHi, false, bf,
                                                     (short)0, acc, false, false);
        acc = __builtin_amdgcn_wmma_f32_16x16x32_f16(false, aLo, false, bf,
                                                     (short)0, acc, false, false);
    }
    // acc[r] = prob[q0 + r + hi8, label[t]]  (exact to ~2^-22 via hi/lo split)

    const float tx0 = tcx - 0.5f * tw, ty0 = tcy - 0.5f * th;
    const float tx1 = tcx + 0.5f * tw, ty1 = tcy + 0.5f * th;
    const float a2  = (tx1 - tx0) * (ty1 - ty0);

    #pragma unroll
    for (int r = 0; r < 8; ++r) {
        const int qrow = r + hi8;              // D layout: M = r + 8*(lane>=16)
        const int q = q0 + qrow;
        if (q < QN) {
            const float pcx = sPB[qrow * 4 + 0], pcy = sPB[qrow * 4 + 1];
            const float pw  = sPB[qrow * 4 + 2], ph  = sPB[qrow * 4 + 3];
            const float px0 = pcx - 0.5f * pw, py0 = pcy - 0.5f * ph;
            const float px1 = pcx + 0.5f * pw, py1 = pcy + 0.5f * ph;
            const float a1  = (px1 - px0) * (py1 - py0);
            const float iw  = fmaxf(fminf(px1, tx1) - fmaxf(px0, tx0), 0.f);
            const float ih  = fmaxf(fminf(py1, ty1) - fmaxf(py0, ty0), 0.f);
            const float inter = iw * ih;
            const float uni   = a1 + a2 - inter;
            const float iou   = inter / uni;
            const float cw  = fmaxf(fmaxf(px1, tx1) - fminf(px0, tx0), 0.f);
            const float ch  = fmaxf(fmaxf(py1, ty1) - fminf(py0, ty0), 0.f);
            const float ac  = cw * ch;
            const float giou = iou - (ac - uni) / ac;
            const float l1 = fabsf(pcx - tcx) + fabsf(pcy - tcy)
                           + fabsf(pw - tw)  + fabsf(ph - th);
            const float cost = -acc[r] + 5.f * l1 - 2.f * giou;
            Cout[((size_t)b * QN + q) * TN + t] = cost;
        }
    }
}

// One block per batch. Incremental greedy: cache per-row (min, argmin) in LDS;
// per step reduce over 900 row-minima, then rescan only rows whose cached
// argmin column was just consumed. Used rows encoded as rowMin == +inf.
__global__ __launch_bounds__(1024) void detr_greedy_kernel(
    const float* __restrict__ Cmat,   // [B,QN,TN]
    float* __restrict__ srcOut,       // [B,TN] (indices as floats)
    float* __restrict__ tgtOut)       // [B,TN]
{
    __shared__ float rowMin[QN];
    __shared__ short rowArg[QN];
    __shared__ unsigned char colUsed[TN];
    __shared__ float redVal[32];
    __shared__ int   redIdx[32];
    __shared__ int   selJ;

    const int b   = blockIdx.x;
    const int tid = threadIdx.x;
    const float* Cb  = Cmat + (size_t)b * QN * TN;
    const float  INF = __builtin_inff();

    if (tid < TN) colUsed[tid] = 0;
    if (tid < QN) {
        const float* row = Cb + (size_t)tid * TN;
        float m = INF; int a = 0;
        for (int j = 0; j < TN; ++j) {
            const float v = row[j];
            if (v < m) { m = v; a = j; }       // strict < keeps smallest j on tie
        }
        rowMin[tid] = m;
        rowArg[tid] = (short)a;
    }
    __syncthreads();

    for (int step = 0; step < TN; ++step) {
        // ---- phase 1: argmin over cached row minima (flat-index tie-break) ----
        float best = (tid < QN) ? rowMin[tid] : INF;
        int   bidx = (tid < QN) ? tid : 0x7fffffff;
        #pragma unroll
        for (int off = 16; off > 0; off >>= 1) {
            const float ov = __shfl_down(best, off);
            const int   oi = __shfl_down(bidx, off);
            if (ov < best || (ov == best && oi < bidx)) { best = ov; bidx = oi; }
        }
        if ((tid & 31) == 0) { redVal[tid >> 5] = best; redIdx[tid >> 5] = bidx; }
        __syncthreads();
        if (tid < 32) {
            best = redVal[tid]; bidx = redIdx[tid];
            #pragma unroll
            for (int off = 16; off > 0; off >>= 1) {
                const float ov = __shfl_down(best, off);
                const int   oi = __shfl_down(bidx, off);
                if (ov < best || (ov == best && oi < bidx)) { best = ov; bidx = oi; }
            }
            if (tid == 0) {
                const int i = bidx;
                const int j = (int)rowArg[i];
                selJ = j;
                colUsed[j] = 1;
                rowMin[i]  = INF;              // retire row i
                srcOut[b * TN + step] = (float)i;
                tgtOut[b * TN + step] = (float)j;
            }
        }
        __syncthreads();

        // ---- phase 2: rescan only rows whose argmin column was consumed ----
        const int j = selJ;
        if (tid < QN && rowMin[tid] != INF && (int)rowArg[tid] == j) {
            const float* row = Cb + (size_t)tid * TN;
            float m = INF; int a = 0;
            for (int c = 0; c < TN; ++c) {
                if (!colUsed[c]) {
                    const float v = row[c];
                    if (v < m) { m = v; a = c; }
                }
            }
            rowMin[tid] = m;
            rowArg[tid] = (short)a;
        }
        __syncthreads();
    }
}

extern "C" void kernel_launch(void* const* d_in, const int* in_sizes, int n_in,
                              void* d_out, int out_size, void* d_ws, size_t ws_size,
                              hipStream_t stream) {
    (void)n_in; (void)out_size; (void)d_ws; (void)ws_size;
    const float* logits = (const float*)d_in[0];   // [B,900,91]
    const float* pboxes = (const float*)d_in[1];   // [B,900,4]
    const int*   labels = (const int*)  d_in[2];   // [B,128]
    const float* tboxes = (const float*)d_in[3];   // [B,128,4]

    const int Bn = in_sizes[2] / TN;

    float* out    = (float*)d_out;
    float* srcOut = out;                               // [B,128]
    float* tgtOut = out + (size_t)Bn * TN;             // [B,128]
    float* Cout   = out + (size_t)2 * Bn * TN;         // [B,900,128]

    dim3 gridA(QT_TILES, Bn);
    detr_cost_kernel<<<gridA, 256, 0, stream>>>(logits, pboxes, labels, tboxes, Cout);
    detr_greedy_kernel<<<Bn, 1024, 0, stream>>>(Cout, srcOut, tgtOut);
}
